// HyperbolicMLR_33045478375870
// MI455X (gfx1250) — matrix operand order repaired and checked by reading, not compiled, not run on hardware
//
#include <hip/hip_runtime.h>
#include <math.h>

// ---------------------------------------------------------------------------
// HyperbolicMLR forward, MI455X (gfx1250, wave32, WMMA).
// B=C=4096, D=1024, k(curvature)=1.
//   px = x_proj . p^T, xa = x_proj . a^T  (two f16->f32 WMMA GEMM chains,
//   fused, sharing the A-operand), then the arcsinh distance epilogue.
// Block tile 128x128, 8 waves as 4(M) x 2(N), wave tile 32x64,
// staged K-depth 64 (two 32-deep WMMA slices per barrier),
// B-fragments double-buffered across tn to hide LDS latency.
// ---------------------------------------------------------------------------

#define NB 4096   // batch rows
#define NC 4096   // classes
#define ND 1024   // feature dim

#define BM 128    // block tile rows (x)
#define BN 128    // block tile cols (classes)
#define BK 64     // staged K depth (2 WMMA K-slices)
#define KPAD 72   // halves per LDS row (pad 64 -> 72: row*36 mod 64 all distinct)

#define EPSF 1e-15f

typedef __attribute__((ext_vector_type(16))) _Float16 v16h;
typedef __attribute__((ext_vector_type(8)))  float    v8f;

union Frag { v16h v; uint4 q[2]; };

// ---------------------------------------------------------------------------
// Kernel 1: projection + f16 conversion + row/class statistics.
// blockIdx.x in [0,NB): x row.  blockIdx.x in [NB,NB+NC): class row.
// ---------------------------------------------------------------------------
__global__ __launch_bounds__(256)
void hmlr_prep(const float* __restrict__ x, const float* __restrict__ a,
               const float* __restrict__ p,
               _Float16* __restrict__ xh, _Float16* __restrict__ ph,
               _Float16* __restrict__ ah,
               float* __restrict__ x2v, float* __restrict__ p2v,
               float* __restrict__ pav, float* __restrict__ anv)
{
    __shared__ float red[3][256];
    const int t = threadIdx.x;

    if (blockIdx.x < NB) {
        const int row = blockIdx.x;
        const float* xr = x + (size_t)row * ND;
        float v[4];
        float ss = 0.f;
        #pragma unroll
        for (int i = 0; i < 4; ++i) { v[i] = xr[t + 256 * i]; ss += v[i] * v[i]; }
        red[0][t] = ss;
        __syncthreads();
        for (int s = 128; s > 0; s >>= 1) {
            if (t < s) red[0][t] += red[0][t + s];
            __syncthreads();
        }
        const float sum = red[0][0];
        const float xn   = fmaxf(sqrtf(sum), EPSF);
        const float maxn = 1.0f - 1e-5f;                 // (1-eps)/sqrt(k), k=1
        const float s    = (xn > maxn) ? (maxn / xn) : 1.0f;
        _Float16* xo = xh + (size_t)row * ND;
        #pragma unroll
        for (int i = 0; i < 4; ++i) xo[t + 256 * i] = (_Float16)(v[i] * s);
        if (t == 0) x2v[row] = sum * s * s;              // ||x_proj||^2
    } else {
        const int c = blockIdx.x - NB;
        const float* pr = p + (size_t)c * ND;
        const float* ar = a + (size_t)c * ND;
        float vp[4], va[4];
        float sp2 = 0.f, spa = 0.f, sa2 = 0.f;
        #pragma unroll
        for (int i = 0; i < 4; ++i) {
            vp[i] = pr[t + 256 * i];
            va[i] = ar[t + 256 * i];
            sp2 += vp[i] * vp[i];
            spa += vp[i] * va[i];
            sa2 += va[i] * va[i];
        }
        red[0][t] = sp2; red[1][t] = spa; red[2][t] = sa2;
        __syncthreads();
        for (int s = 128; s > 0; s >>= 1) {
            if (t < s) {
                red[0][t] += red[0][t + s];
                red[1][t] += red[1][t + s];
                red[2][t] += red[2][t + s];
            }
            __syncthreads();
        }
        _Float16* po = ph + (size_t)c * ND;
        _Float16* ao = ah + (size_t)c * ND;
        #pragma unroll
        for (int i = 0; i < 4; ++i) {
            po[t + 256 * i] = (_Float16)vp[i];
            ao[t + 256 * i] = (_Float16)va[i];
        }
        if (t == 0) {
            p2v[c] = red[0][0];
            pav[c] = red[1][0];
            anv[c] = sqrtf(red[2][0]);
        }
    }
}

// ---------------------------------------------------------------------------
// Kernel 2: fused dual-GEMM (px & xa, shared A operand) + hyperbolic epilogue.
// ---------------------------------------------------------------------------
__global__ __launch_bounds__(256)
void hmlr_gemm(const _Float16* __restrict__ xh, const _Float16* __restrict__ ph,
               const _Float16* __restrict__ ah,
               const float* __restrict__ x2v, const float* __restrict__ p2v,
               const float* __restrict__ pav, const float* __restrict__ anv,
               float* __restrict__ out)
{
    __shared__ __align__(16) _Float16 xs[BM * KPAD];
    __shared__ __align__(16) _Float16 ps[BN * KPAD];
    __shared__ __align__(16) _Float16 as2[BN * KPAD];

    const int t     = threadIdx.x;
    const int lane  = t & 31;
    const int wave  = t >> 5;
    const int waveM = wave & 3;   // 0..3 -> 32-row group
    const int waveN = wave >> 2;  // 0..1 -> 64-col group
    const int rowBase = blockIdx.y * BM;
    const int colBase = blockIdx.x * BN;

    const v8f vzero = {0.f, 0.f, 0.f, 0.f, 0.f, 0.f, 0.f, 0.f};
    v8f accP[2][4], accA[2][4];
    #pragma unroll
    for (int i = 0; i < 2; ++i)
        #pragma unroll
        for (int j = 0; j < 4; ++j) { accP[i][j] = vzero; accA[i][j] = vzero; }

    const int half = lane >> 4;   // which 16-lane half of the wave
    const int lr   = lane & 15;

    for (int k0 = 0; k0 < ND; k0 += BK) {
        __syncthreads();
        // ---- stage 128x64 (x) + 2x 128x64 (p,a) f16 tiles into LDS ----
        // Each tile = 1024 uint4 chunks; 256 threads -> 4 chunks per tile.
        #pragma unroll
        for (int r = 0; r < 4; ++r) {
            const int idx = t + 256 * r;
            const int row = idx >> 3, kc = idx & 7;
            const size_t gx = (size_t)(rowBase + row) * ND + k0 + kc * 8;
            const size_t gc = (size_t)(colBase + row) * ND + k0 + kc * 8;
            const uint4 dx = *reinterpret_cast<const uint4*>(xh + gx);
            const uint4 dp = *reinterpret_cast<const uint4*>(ph + gc);
            const uint4 da = *reinterpret_cast<const uint4*>(ah + gc);
            const int loff = row * (KPAD * 2) + kc * 16;
            *reinterpret_cast<uint4*>((char*)xs  + loff) = dx;
            *reinterpret_cast<uint4*>((char*)ps  + loff) = dp;
            *reinterpret_cast<uint4*>((char*)as2 + loff) = da;
        }
        __syncthreads();

        // ---- two 32-deep WMMA K-slices per stage ----
        #pragma unroll
        for (int ks = 0; ks < 2; ++ks) {
            const int kb = ks * 64;   // byte offset of K-slice within LDS row

            // A fragments (16x32 f16): lane = row, contiguous chunk at K h*16
            Frag afrag[2];
            #pragma unroll
            for (int tm = 0; tm < 2; ++tm) {
                const int row = waveM * 32 + tm * 16 + lr;
                const char* base = (const char*)xs + row * (KPAD * 2) + kb;
                afrag[tm].q[0] = *reinterpret_cast<const uint4*>(base + half * 32);
                afrag[tm].q[1] = *reinterpret_cast<const uint4*>(base + half * 32 + 16);
            }

            // B fragments (32x16 f16): lane = col, contiguous chunk at K h*16.
            // Double-buffered across tn so each WMMA pair waits on loads
            // issued a full tn-iteration earlier.
            Frag bpb[2], bab[2];
            {
                const int col = waveN * 64 + lr;
                const char* pb = (const char*)ps  + col * (KPAD * 2) + kb + half * 32;
                const char* ab = (const char*)as2 + col * (KPAD * 2) + kb + half * 32;
                bpb[0].q[0] = *reinterpret_cast<const uint4*>(pb);
                bpb[0].q[1] = *reinterpret_cast<const uint4*>(pb + 16);
                bab[0].q[0] = *reinterpret_cast<const uint4*>(ab);
                bab[0].q[1] = *reinterpret_cast<const uint4*>(ab + 16);
            }
            #pragma unroll
            for (int tn = 0; tn < 4; ++tn) {
                const int cur = tn & 1;
                const int nxt = cur ^ 1;
                if (tn < 3) {
                    const int col = waveN * 64 + (tn + 1) * 16 + lr;
                    const char* pb = (const char*)ps  + col * (KPAD * 2) + kb + half * 32;
                    const char* ab = (const char*)as2 + col * (KPAD * 2) + kb + half * 32;
                    bpb[nxt].q[0] = *reinterpret_cast<const uint4*>(pb);
                    bpb[nxt].q[1] = *reinterpret_cast<const uint4*>(pb + 16);
                    bab[nxt].q[0] = *reinterpret_cast<const uint4*>(ab);
                    bab[nxt].q[1] = *reinterpret_cast<const uint4*>(ab + 16);
                }
                #pragma unroll
                for (int tm = 0; tm < 2; ++tm) {
                    accP[tm][tn] = __builtin_amdgcn_wmma_f32_16x16x32_f16(
                        false, afrag[tm].v, false, bpb[cur].v, (short)0,
                        accP[tm][tn], false, false);
                    accA[tm][tn] = __builtin_amdgcn_wmma_f32_16x16x32_f16(
                        false, afrag[tm].v, false, bab[cur].v, (short)0,
                        accA[tm][tn], false, false);
                }
            }
        }
    }

    // ---- hyperbolic-distance epilogue (k = 1) -------------------------------
    // C/D layout: lanes 0-15 -> N=lane, VGPR j -> M=j; lanes 16-31 -> M=j+8.
    #pragma unroll
    for (int tm = 0; tm < 2; ++tm) {
        const int row0 = rowBase + waveM * 32 + tm * 16 + half * 8;
        #pragma unroll
        for (int tn = 0; tn < 4; ++tn) {
            const int col = colBase + waveN * 64 + tn * 16 + lr;
            const float p2c = p2v[col];
            const float pac = pav[col];
            const float anc = anv[col];
            const float Bc  = 1.f - p2c;
            #pragma unroll
            for (int j = 0; j < 8; ++j) {
                const int   row = row0 + j;
                const float x2r = x2v[row];
                const float px  = accP[tm][tn][j];
                const float xa  = accA[tm][tn][j];
                const float A   = 1.f - 2.f * px + x2r;
                const float den = fmaxf(1.f - 2.f * px + x2r * p2c, EPSF);
                const float r   = 1.f / den;
                const float dn2 = fmaxf(
                    (A * A * p2c + Bc * Bc * x2r - 2.f * A * Bc * px) * r * r, EPSF);
                const float sc  = (-A * pac + Bc * xa) * r;
                // d0 = (1+dn2)*||a|| > 0, so clamp_abs == fmaxf
                const float d0  = fmaxf((1.f + dn2) * anc, EPSF);
                const float z   = 2.f * sc / d0;
                // asinh(z) = sign(z) * log(|z| + sqrt(z^2+1)), arg >= 1: safe
                const float az  = fabsf(z);
                const float dist = copysignf(logf(az + sqrtf(az * az + 1.f)), z);
                out[(size_t)row * NC + col] = -dist;
            }
        }
    }
}

// ---------------------------------------------------------------------------
extern "C" void kernel_launch(void* const* d_in, const int* in_sizes, int n_in,
                              void* d_out, int out_size, void* d_ws, size_t ws_size,
                              hipStream_t stream) {
    const float* x = (const float*)d_in[0];  // [4096,1024]
    const float* a = (const float*)d_in[1];  // [4096,1024]
    const float* p = (const float*)d_in[2];  // [4096,1024]
    float* out = (float*)d_out;              // [4096,4096]

    // Workspace layout (~24.1 MB): f16 copies + per-row/class stats.
    char* ws = (char*)d_ws;
    _Float16* xh = (_Float16*)(ws);
    _Float16* ph = (_Float16*)(ws + (size_t)8  * 1024 * 1024);
    _Float16* ah = (_Float16*)(ws + (size_t)16 * 1024 * 1024);
    float* x2v = (float*)(ws + (size_t)24 * 1024 * 1024);
    float* p2v = x2v + NB;
    float* pav = p2v + NC;
    float* anv = pav + NC;

    hmlr_prep<<<NB + NC, 256, 0, stream>>>(x, a, p, xh, ph, ah,
                                           x2v, p2v, pav, anv);

    dim3 grid(NC / BN, NB / BM);   // 32 x 32 tiles
    hmlr_gemm<<<grid, 256, 0, stream>>>(xh, ph, ah, x2v, p2v, pav, anv, out);
}
